// GATv2NBNet_8564164788937
// MI455X (gfx1250) — compile-verified, matrix operand order checked by reading, not run on hardware
//
#include <hip/hip_runtime.h>
#include <hip/hip_fp16.h>
#include <cstdint>

typedef __attribute__((ext_vector_type(16))) _Float16     v16h;
typedef __attribute__((ext_vector_type(8)))  float        v8f;
typedef __attribute__((ext_vector_type(4)))  unsigned int u32x4;
typedef __attribute__((ext_vector_type(8)))  unsigned int u32x8;

#define N_NODES 50000
#define N_EDGES 800000
#define F_DIM   128
#define H_DIM   64
#define IN_DIM  129
#define M_TOT   (N_EDGES + N_NODES)
#define NEG_SLOPE 0.2f

// ---------------------------------------------------------------------------
// Tensor Data Mover: async 2D f16 tile load global -> LDS.
// D# built per CDNA5 ISA §8: group0 = {count=1, lds_addr, global_addr[56:0],
// type=2}; group1 = {data_size=1(2B), tensor_dim0/1, tile_dim0/1, dim0_stride}.
// 2-group asm form (VADDR2/VADDR3 = NULL) is valid for tensors up to 2D.
// Tracked by TENSORcnt; wave issuing it must s_wait_tensorcnt before use.
// ---------------------------------------------------------------------------
__device__ __forceinline__ void tdm_load_2d_f16(const _Float16* gsrc,
                                                unsigned lds_byte_off,
                                                unsigned rows, unsigned cols) {
    unsigned long long ga = (unsigned long long)(uintptr_t)gsrc;
    u32x4 g0;
    g0[0] = 1u;                                            // count=1, user desc
    g0[1] = lds_byte_off;                                  // LDS dest
    g0[2] = (unsigned)(ga & 0xffffffffu);                  // global addr lo
    g0[3] = (unsigned)((ga >> 32) & 0x01ffffffu) | (2u << 30); // addr hi | type=2
    u32x8 g1;
    g1[0] = (1u << 16);                                    // data_size=1 (2 bytes)
    g1[1] = (cols & 0xffffu) << 16;                        // tensor_dim0 lo16
    g1[2] = (cols >> 16) | ((rows & 0xffffu) << 16);       // td0 hi | tensor_dim1 lo
    g1[3] = (rows >> 16) | ((cols & 0xffffu) << 16);       // td1 hi | tile_dim0
    g1[4] = (rows & 0xffffu);                              // tile_dim1 (tile_dim2=0)
    g1[5] = cols;                                          // tensor_dim0_stride lo32
    g1[6] = 0u;                                            // stride hi | d1stride lo
    g1[7] = 0u;                                            // d1stride hi
    asm volatile("tensor_load_to_lds %0, %1" :: "s"(g0), "s"(g1) : "memory");
}

__device__ __forceinline__ unsigned lds_off_u32(const void* p) {
    // LDS aperture: ADDR[31:0] is the LDS byte offset (ISA §10.2)
    return (unsigned)(uintptr_t)p;
}

// ---------------------------------------------------------------------------
// float atomic max via signed/unsigned integer ordering trick
// (init value must be -inf = 0xff800000)
// ---------------------------------------------------------------------------
__device__ __forceinline__ void atomicMaxFloat(float* addr, float v) {
    if (v >= 0.0f) atomicMax((int*)addr, __float_as_int(v));
    else           atomicMin((unsigned int*)addr, __float_as_uint(v));
}

// ---------------------------------------------------------------------------
// 0a) zero/neg-inf init of softmax state (ws is poisoned 0xAA by the harness)
// ---------------------------------------------------------------------------
__global__ __launch_bounds__(256) void init_kernel(float* __restrict__ emax,
                                                   float* __restrict__ denom,
                                                   float* __restrict__ accum) {
    int i = blockIdx.x * 256 + threadIdx.x;
    if (i < N_NODES) emax[i] = -__builtin_inff();
    int j = i - N_NODES;
    if (j >= 0 && j < N_NODES) denom[j] = 0.0f;
    int a = i - 2 * N_NODES;
    if (a >= 0 && a < N_NODES * H_DIM) accum[a] = 0.0f;
}

// ---------------------------------------------------------------------------
// 0b) one-shot weight conversion f32 -> f16 (drops Wl/Wr binary column;
//     that column is applied as an fp32 rank-1 fix in proj_kernel)
// ---------------------------------------------------------------------------
__global__ __launch_bounds__(256) void convert_weights_kernel(
    const float* __restrict__ Wl, const float* __restrict__ Wr,
    const float* __restrict__ Wo,
    _Float16* __restrict__ Wl16, _Float16* __restrict__ Wr16,
    _Float16* __restrict__ Wo16) {
    int i = blockIdx.x * 256 + threadIdx.x;
    if (i < H_DIM * F_DIM) {                 // [64][128], strip col 128
        int h = i >> 7, k = i & 127;
        Wl16[i] = (_Float16)Wl[h * IN_DIM + k];
        Wr16[i] = (_Float16)Wr[h * IN_DIM + k];
    }
    if (i < F_DIM * H_DIM)                   // [128][64], contiguous
        Wo16[i] = (_Float16)Wo[i];
}

// ---------------------------------------------------------------------------
// 1) node projections: xl = x @ Wl^T + bl, xr = x @ Wr^T + br
//    TDM stages Wl16/Wr16 into LDS while the block stages the x tile;
//    K=128 via 4 WMMA k-steps; binary column (k=128) as fp32 rank-1 fix.
//    block = 128 threads = 4 waves, block = 16 nodes, wave w -> h cols
//    [16w,16w+16)
// ---------------------------------------------------------------------------
__global__ __launch_bounds__(128) void proj_kernel(const float* __restrict__ xf,
                                                   const float* __restrict__ xb,
                                                   const _Float16* __restrict__ Wl16,
                                                   const _Float16* __restrict__ Wr16,
                                                   const float* __restrict__ Wl,
                                                   const float* __restrict__ bl,
                                                   const float* __restrict__ Wr,
                                                   const float* __restrict__ br,
                                                   float* __restrict__ xl_out,
                                                   float* __restrict__ xr_out) {
    __shared__ _Float16 xs[16][F_DIM];             //  4 KB
    __shared__ _Float16 wls[H_DIM * F_DIM];        // 16 KB
    __shared__ _Float16 wrs[H_DIM * F_DIM];        // 16 KB
    __shared__ float    xbin[16];
    const int node0 = blockIdx.x * 16;
    const int tid   = threadIdx.x;

    // wave 0 kicks off async weight DMA (EXEC-independent, once per wave)
    if (tid < 32) {
        tdm_load_2d_f16(Wl16, lds_off_u32(wls), H_DIM, F_DIM);
        tdm_load_2d_f16(Wr16, lds_off_u32(wrs), H_DIM, F_DIM);
    }

    // meanwhile: stage 16x128 fp32 -> f16 x tile into LDS (coalesced)
    for (int i = tid; i < 16 * F_DIM; i += 128) {
        int m = i >> 7, k = i & 127;
        xs[m][k] = (_Float16)xf[(node0 + m) * F_DIM + k];
    }
    if (tid < 16) xbin[tid] = xb[node0 + tid];

    if (tid < 32) __builtin_amdgcn_s_wait_tensorcnt(0);
    __syncthreads();

    const int wave = tid >> 5;
    const int lane = tid & 31;
    const int mrow = lane & 15;
    const int kb   = (lane < 16) ? 0 : 8;
    const int moff = (lane < 16) ? 0 : 8;
    const int hcol = wave * 16 + (lane & 15);

    v8f accl = {};
    v8f accr = {};
#pragma unroll
    for (int kt = 0; kt < 4; ++kt) {
        v16h afrag, bL, bR;
#pragma unroll
        for (int e = 0; e < 16; ++e) {
            int k    = kt * 32 + kb + ((e & 8) ? 16 : 0) + (e & 7);
            afrag[e] = xs[mrow][k];
            bL[e]    = wls[hcol * F_DIM + k];
            bR[e]    = wrs[hcol * F_DIM + k];
        }
        accl = __builtin_amdgcn_wmma_f32_16x16x32_f16(false, afrag, false, bL,
                                                      (short)0, accl, false, false);
        accr = __builtin_amdgcn_wmma_f32_16x16x32_f16(false, afrag, false, bR,
                                                      (short)0, accr, false, false);
    }

    const float wlb = Wl[hcol * IN_DIM + 128];
    const float wrb = Wr[hcol * IN_DIM + 128];
    const float bLv = bl[hcol];
    const float bRv = br[hcol];
#pragma unroll
    for (int r = 0; r < 8; ++r) {
        int   node = node0 + r + moff;
        float xbv  = xbin[r + moff];
        xl_out[node * H_DIM + hcol] = accl[r] + bLv + xbv * wlb;
        xr_out[node * H_DIM + hcol] = accr[r] + bRv + xbv * wrb;
    }
}

// ---------------------------------------------------------------------------
// 2) per-edge GATv2 score e = att . leakyrelu(xl[src]+xr[dst]); atomic segmax
// ---------------------------------------------------------------------------
__global__ __launch_bounds__(256) void edge_score_kernel(const int* __restrict__ ei,
                                                         const float* __restrict__ xl,
                                                         const float* __restrict__ xr,
                                                         const float* __restrict__ att,
                                                         float* __restrict__ evals,
                                                         float* __restrict__ emax) {
    int idx = blockIdx.x * 256 + threadIdx.x;
    if (idx >= M_TOT) return;
    int s, d;
    if (idx < N_EDGES) { s = ei[idx]; d = ei[N_EDGES + idx]; }
    else               { s = d = idx - N_EDGES; }

    const float4* pl = (const float4*)(xl + (size_t)s * H_DIM);
    const float4* pr = (const float4*)(xr + (size_t)d * H_DIM);
    const float4* pa = (const float4*)att;
    float e = 0.0f;
#pragma unroll
    for (int q = 0; q < H_DIM / 4; ++q) {
        float4 a = pl[q], b = pr[q], w = pa[q];
        float mx = a.x + b.x, my = a.y + b.y, mz = a.z + b.z, mw = a.w + b.w;
        mx = mx > 0.0f ? mx : NEG_SLOPE * mx;
        my = my > 0.0f ? my : NEG_SLOPE * my;
        mz = mz > 0.0f ? mz : NEG_SLOPE * mz;
        mw = mw > 0.0f ? mw : NEG_SLOPE * mw;
        e = fmaf(mx, w.x, fmaf(my, w.y, fmaf(mz, w.z, fmaf(mw, w.w, e))));
    }
    evals[idx] = e;
    atomicMaxFloat(&emax[d], e);
}

// ---------------------------------------------------------------------------
// 3) ex = exp(e - emax[dst]); denom += ex; accum[dst] += ex * xl[src]
//    (dividing accum by denom later == softmax-weighted aggregation)
// ---------------------------------------------------------------------------
__global__ __launch_bounds__(256) void edge_scatter_kernel(const int* __restrict__ ei,
                                                           const float* __restrict__ xl,
                                                           const float* __restrict__ evals,
                                                           const float* __restrict__ emax,
                                                           float* __restrict__ denom,
                                                           float* __restrict__ accum) {
    int idx = blockIdx.x * 256 + threadIdx.x;
    if (idx >= M_TOT) return;
    int s, d;
    if (idx < N_EDGES) { s = ei[idx]; d = ei[N_EDGES + idx]; }
    else               { s = d = idx - N_EDGES; }

    float ex = __expf(evals[idx] - emax[d]);
    atomicAdd(&denom[d], ex);

    const float4* pl  = (const float4*)(xl + (size_t)s * H_DIM);
    float*        acc = accum + (size_t)d * H_DIM;
#pragma unroll
    for (int q = 0; q < H_DIM / 4; ++q) {
        float4 v = pl[q];
        atomicAdd(acc + 4 * q + 0, ex * v.x);
        atomicAdd(acc + 4 * q + 1, ex * v.y);
        atomicAdd(acc + 4 * q + 2, ex * v.z);
        atomicAdd(acc + 4 * q + 3, ex * v.w);
    }
}

// ---------------------------------------------------------------------------
// 4) finalize: h = elu(accum/denom + gat_bias); logits = h @ Wo^T + bo (WMMA)
//    TDM stages Wo16 into LDS; block = 16 nodes, 4 waves; wave w -> f cols
//    [32w, 32w+32) as two 16-wide tiles, K=64 = 2 k-steps.
// ---------------------------------------------------------------------------
__global__ __launch_bounds__(128) void out_kernel(const float* __restrict__ accum,
                                                  const float* __restrict__ denom,
                                                  const float* __restrict__ gat_bias,
                                                  const _Float16* __restrict__ Wo16,
                                                  const float* __restrict__ bo,
                                                  float* __restrict__ out) {
    __shared__ _Float16 hs[16][H_DIM];             //  2 KB
    __shared__ _Float16 wos[F_DIM * H_DIM];        // 16 KB
    const int node0 = blockIdx.x * 16;
    const int tid   = threadIdx.x;

    if (tid < 32) tdm_load_2d_f16(Wo16, lds_off_u32(wos), F_DIM, H_DIM);

    for (int i = tid; i < 16 * H_DIM; i += 128) {
        int   m = i >> 6, h = i & 63;
        float v = accum[(node0 + m) * H_DIM + h] / denom[node0 + m] + gat_bias[h];
        v = v > 0.0f ? v : (__expf(v) - 1.0f);   // ELU(alpha=1)
        hs[m][h] = (_Float16)v;
    }

    if (tid < 32) __builtin_amdgcn_s_wait_tensorcnt(0);
    __syncthreads();

    const int wave = tid >> 5;
    const int lane = tid & 31;
    const int mrow = lane & 15;
    const int kb   = (lane < 16) ? 0 : 8;
    const int moff = (lane < 16) ? 0 : 8;

#pragma unroll
    for (int t = 0; t < 2; ++t) {
        const int fcol = wave * 32 + t * 16 + (lane & 15);
        v8f acc = {};
#pragma unroll
        for (int kt = 0; kt < 2; ++kt) {
            v16h afrag, bfrag;
#pragma unroll
            for (int e = 0; e < 16; ++e) {
                int k    = kt * 32 + kb + ((e & 8) ? 16 : 0) + (e & 7);
                afrag[e] = hs[mrow][k];
                bfrag[e] = wos[fcol * H_DIM + k];
            }
            acc = __builtin_amdgcn_wmma_f32_16x16x32_f16(false, afrag, false, bfrag,
                                                         (short)0, acc, false, false);
        }
        const float bias = bo[fcol];
#pragma unroll
        for (int r = 0; r < 8; ++r)
            out[(size_t)(node0 + r + moff) * F_DIM + fcol] = acc[r] + bias;
    }
}

// ---------------------------------------------------------------------------
// 5) softplus(dispersion) -> tail of d_out
// ---------------------------------------------------------------------------
__global__ __launch_bounds__(128) void softplus_kernel(const float* __restrict__ disp,
                                                       float* __restrict__ out_tail) {
    int i = threadIdx.x;
    if (i < F_DIM) {
        float x = disp[i];
        out_tail[i] = (x > 20.0f) ? x : log1pf(__expf(x));
    }
}

// ---------------------------------------------------------------------------
extern "C" void kernel_launch(void* const* d_in, const int* in_sizes, int n_in,
                              void* d_out, int out_size, void* d_ws, size_t ws_size,
                              hipStream_t stream) {
    const float* xf   = (const float*)d_in[0];   // [N,128]
    const float* xb   = (const float*)d_in[1];   // [N]
    const int*   ei   = (const int*)d_in[2];     // [2,E]
    const float* Wl   = (const float*)d_in[3];   // [64,129]
    const float* bl   = (const float*)d_in[4];   // [64]
    const float* Wr   = (const float*)d_in[5];   // [64,129]
    const float* br   = (const float*)d_in[6];   // [64]
    const float* att  = (const float*)d_in[7];   // [64]
    const float* gb   = (const float*)d_in[8];   // [64]
    const float* Wo   = (const float*)d_in[9];   // [128,64]
    const float* bo   = (const float*)d_in[10];  // [128]
    const float* disp = (const float*)d_in[11];  // [128]
    float* out = (float*)d_out;                  // [N*128 + 128]

    // workspace carve
    float* ws    = (float*)d_ws;
    float* xl    = ws;                                   // N*64 f32
    float* xr    = xl + (size_t)N_NODES * H_DIM;         // N*64
    float* accum = xr + (size_t)N_NODES * H_DIM;         // N*64
    float* evals = accum + (size_t)N_NODES * H_DIM;      // M_TOT
    float* emax  = evals + (size_t)M_TOT;                // N
    float* denom = emax + N_NODES;                       // N
    _Float16* Wl16 = (_Float16*)(denom + N_NODES);       // 64*128 f16
    _Float16* Wr16 = Wl16 + H_DIM * F_DIM;               // 64*128 f16
    _Float16* Wo16 = Wr16 + H_DIM * F_DIM;               // 128*64 f16

    const int initN = 2 * N_NODES + N_NODES * H_DIM;
    init_kernel<<<(initN + 255) / 256, 256, 0, stream>>>(emax, denom, accum);

    convert_weights_kernel<<<(H_DIM * F_DIM + 255) / 256, 256, 0, stream>>>(
        Wl, Wr, Wo, Wl16, Wr16, Wo16);

    proj_kernel<<<N_NODES / 16, 128, 0, stream>>>(xf, xb, Wl16, Wr16, Wl, bl,
                                                  Wr, br, xl, xr);

    edge_score_kernel<<<(M_TOT + 255) / 256, 256, 0, stream>>>(ei, xl, xr, att,
                                                               evals, emax);

    edge_scatter_kernel<<<(M_TOT + 255) / 256, 256, 0, stream>>>(ei, xl, evals,
                                                                 emax, denom, accum);

    out_kernel<<<N_NODES / 16, 128, 0, stream>>>(accum, denom, gb, Wo16, bo, out);

    softplus_kernel<<<1, 128, 0, stream>>>(disp, out + (size_t)N_NODES * F_DIM);
}
